// MHSS_SSSM_Block_12232066859080
// MI455X (gfx1250) — compile-verified, matrix operand order checked by reading, not run on hardware
//
#include <hip/hip_runtime.h>
#include <hip/hip_bf16.h>

typedef __attribute__((ext_vector_type(16))) _Float16 v16h;
typedef __attribute__((ext_vector_type(8)))  float    v8f;
typedef __attribute__((ext_vector_type(4)))  unsigned v4u;
typedef int v4i_ __attribute__((vector_size(16)));          // matches builtin proto

#define NDIR    2
#define BATCH   8
#define SEQ     1024          // H*W = 32*32
#define DM      512           // d_model
#define DI      512           // d_inner
#define DS      16            // d_state
#define DCONV   4
#define DTR     32            // dt_rank
#define MROWS   (BATCH*SEQ)   // 8192 rows per direction

// GEMM block tile
#define GBM 128
#define GBN 64
#define GBK 32

static __device__ __forceinline__ float sigmoidf_(float x) { return 1.0f / (1.0f + __expf(-x)); }
static __device__ __forceinline__ float siluf_(float x)    { return x * sigmoidf_(x); }

__device__ __forceinline__ v8f wmma16x16x32(v16h a, v16h b, v8f c) {
  // D = A(16x32 f16) * B(32x16 f16) + C(16x16 f32)
  return __builtin_amdgcn_wmma_f32_16x16x32_f16(
      /*neg_a=*/false, a, /*neg_b=*/false, b,
      /*c_mod=*/(short)0, c, /*reuse_a=*/false, /*reuse_b=*/false);
}

// ---- async global->LDS (16B per lane), guarded for toolchain portability ----
#if __has_builtin(__builtin_amdgcn_global_load_async_to_lds_b128)
#define HAVE_ASYNC_LDS 1
typedef __attribute__((address_space(1))) v4i_* g4p_t;
typedef __attribute__((address_space(3))) v4i_* l4p_t;
__device__ __forceinline__ void cp16_g2l(const _Float16* g, _Float16* l) {
  __builtin_amdgcn_global_load_async_to_lds_b128(
      (g4p_t)(size_t)g,                 // int4 addrspace(1)*
      (l4p_t)(unsigned)(size_t)l,       // int4 addrspace(3)*  (low 32b = LDS offset)
      0, 0);
}
#else
#define HAVE_ASYNC_LDS 0
__device__ __forceinline__ void cp16_g2l(const _Float16* g, _Float16* l) {
  *(v4u*)l = *(const v4u*)g;   // global_load_b128 + ds_store_b128
}
#endif

#if __has_builtin(__builtin_amdgcn_s_wait_asynccnt)
#define WAIT_ASYNC(n) __builtin_amdgcn_s_wait_asynccnt((unsigned short)(n))
#elif HAVE_ASYNC_LDS
#define WAIT_ASYNC(n) asm volatile("s_wait_asynccnt %0" ::"n"(n) : "memory")
#else
#define WAIT_ASYNC(n) ((void)0)
#endif

// ---------------------------------------------------------------------------
// f32 -> f16 conversion (weights)
// ---------------------------------------------------------------------------
__global__ void k_f32_to_f16(const float* __restrict__ in, _Float16* __restrict__ out, int n) {
  int i = blockIdx.x * blockDim.x + threadIdx.x;
  if (i < n) out[i] = (_Float16)in[i];
}

// ---------------------------------------------------------------------------
// Build (dir, b, l, c) f16 sequences from NCHW x; dir 1 is flipped along L.
// ---------------------------------------------------------------------------
__global__ void k_xseq(const float* __restrict__ x, _Float16* __restrict__ xseq) {
  unsigned i = blockIdx.x * blockDim.x + threadIdx.x;    // 2*8*1024*512 = 2^23
  if (i >= (unsigned)(NDIR * MROWS * DM)) return;
  unsigned c   = i & (DM - 1);
  unsigned l   = (i >> 9) & (SEQ - 1);
  unsigned b   = (i >> 19) & (BATCH - 1);
  unsigned dir = i >> 22;
  unsigned ls  = dir ? (SEQ - 1u - l) : l;
  float v = x[((size_t)(b * DM + c)) * SEQ + ls];
  xseq[((size_t)(dir * MROWS + b * SEQ + l)) * DM + c] = (_Float16)v;
}

// ---------------------------------------------------------------------------
// Block-cooperative WMMA GEMM:  D[m,n] = sum_k A[m*K+k] * W[n*K+k]
// Block (8 waves) computes a 128x64 tile; each wave a 32x32 sub-tile.
// A/B K-slices are staged in LDS with double-buffered async global->LDS
// copies (ASYNCcnt), then consumed via ds_load -> v_wmma_f32_16x16x32_f16.
// ---------------------------------------------------------------------------
__global__ __launch_bounds__(256) void k_gemm_wmma(const _Float16* __restrict__ A,
                                                   const _Float16* __restrict__ W,
                                                   float* __restrict__ D,
                                                   int M, int N, int K) {
  __shared__ __align__(64) _Float16 lA[2][GBM * GBK];   // 8 KB per buffer
  __shared__ __align__(64) _Float16 lB[2][GBN * GBK];   // 4 KB per buffer

  const int tid  = threadIdx.x;
  const int lane = tid & 31;
  const int wave = tid >> 5;
  const int nt   = N / GBN;
  const int tm0  = (blockIdx.x / nt) * GBM;
  const int tn0  = (blockIdx.x % nt) * GBN;
  const int wm   = wave >> 1;          // 0..3 -> M sub-tile
  const int wn   = wave & 1;           // 0..1 -> N sub-tile
  const int half = lane >> 4;
  const int r    = lane & 15;

  // per-thread copy slots: A = 512 16B chunks (2/thread), B = 256 (1/thread)
  const int ar0 = (tid + 0)   >> 2, ac0 = ((tid + 0)   & 3) * 8;
  const int ar1 = (tid + 256) >> 2, ac1 = ((tid + 256) & 3) * 8;
  const int br  = tid >> 2,         bc  = (tid & 3) * 8;

  auto stage = [&](int buf, int kk) {
    cp16_g2l(A + (size_t)(tm0 + ar0) * K + kk + ac0, &lA[buf][ar0 * GBK + ac0]);
    cp16_g2l(A + (size_t)(tm0 + ar1) * K + kk + ac1, &lA[buf][ar1 * GBK + ac1]);
    cp16_g2l(W + (size_t)(tn0 + br)  * K + kk + bc,  &lB[buf][br  * GBK + bc]);
  };

  v8f acc00 = {}, acc01 = {}, acc10 = {}, acc11 = {};
  stage(0, 0);
  int buf = 0;
  for (int kk = 0; kk < K; kk += GBK) {
    if (kk + GBK < K) {
      stage(buf ^ 1, kk + GBK);   // prefetch next slice into other buffer
      WAIT_ASYNC(3);              // previous stage (3 async ops) has landed
    } else {
      WAIT_ASYNC(0);
    }
    __syncthreads();

    const _Float16* la = &lA[buf][(wm * 32 + r) * GBK + half * 16];
    const _Float16* lb = &lB[buf][(wn * 32 + r) * GBK + half * 16];
    v16h a0 = *(const v16h*)la;
    v16h a1 = *(const v16h*)(la + 16 * GBK);
    v16h b0 = *(const v16h*)lb;
    v16h b1 = *(const v16h*)(lb + 16 * GBK);
    acc00 = wmma16x16x32(a0, b0, acc00);
    acc01 = wmma16x16x32(a0, b1, acc01);
    acc10 = wmma16x16x32(a1, b0, acc10);
    acc11 = wmma16x16x32(a1, b1, acc11);

    __syncthreads();              // protect buffer about to be overwritten
    buf ^= 1;
  }

  // C/D layout: VGPR i holds row (i + 8*half); lane&15 is the column.
  const int tm = tm0 + wm * 32;
  const int tn = tn0 + wn * 32;
#pragma unroll
  for (int i = 0; i < 8; ++i) {
    int row = tm + i + half * 8;
    int col = tn + r;
    D[(size_t)row * N + col]             = acc00[i];
    D[(size_t)row * N + col + 16]        = acc01[i];
    D[(size_t)(row + 16) * N + col]      = acc10[i];
    D[(size_t)(row + 16) * N + col + 16] = acc11[i];
  }
}

// ---------------------------------------------------------------------------
// Causal depthwise conv (width 4) + bias + SiLU.  xz rows are 1024 wide
// (xin = cols [0,512)).  Writes u as f32 and f16.
// ---------------------------------------------------------------------------
__global__ void k_conv_silu(const float* __restrict__ xz,
                            const float* __restrict__ cw,   // (512,4)
                            const float* __restrict__ cb,   // (512)
                            float* __restrict__ u32,
                            _Float16* __restrict__ u16) {
  unsigned i = blockIdx.x * blockDim.x + threadIdx.x;       // 8192*512
  if (i >= (unsigned)(MROWS * DI)) return;
  unsigned c = i & (DI - 1);
  unsigned l = (i >> 9) & (SEQ - 1);
  unsigned b = i >> 19;
  float acc = cb[c];
#pragma unroll
  for (int j = 0; j < DCONV; ++j) {
    int ll = (int)l - (DCONV - 1) + j;
    if (ll >= 0)
      acc += cw[c * DCONV + j] * xz[((size_t)(b * SEQ + ll)) * (2 * DI) + c];
  }
  float u = siluf_(acc);
  size_t o = (size_t)(b * SEQ + l) * DI + c;
  u32[o] = u;
  u16[o] = (_Float16)u;
}

// ---------------------------------------------------------------------------
// Extract dt_rank columns of proj (f32, row stride 64) as f16 GEMM input.
// ---------------------------------------------------------------------------
__global__ void k_dtsplit(const float* __restrict__ proj, _Float16* __restrict__ dt16) {
  unsigned i = blockIdx.x * blockDim.x + threadIdx.x;       // 8192*32
  if (i >= (unsigned)(MROWS * DTR)) return;
  unsigned r = i & (DTR - 1);
  unsigned m = i >> 5;
  dt16[(size_t)m * DTR + r] = (_Float16)proj[(size_t)m * (DTR + 2 * DS) + r];
}

// ---------------------------------------------------------------------------
// dt = softplus(dt_raw + bias)   (in place)
// ---------------------------------------------------------------------------
__global__ void k_softplus(float* __restrict__ dt, const float* __restrict__ bias) {
  unsigned i = blockIdx.x * blockDim.x + threadIdx.x;       // 8192*512
  if (i >= (unsigned)(MROWS * DI)) return;
  unsigned c = i & (DI - 1);
  float v = dt[i] + bias[c];
  dt[i] = (v > 20.0f) ? v : log1pf(__expf(v));
}

// ---------------------------------------------------------------------------
// Selective scan.  One thread per (dir, b, channel); n=16 states in regs.
//   h = exp(dt*A)*h + dt*B*u ;  y = <h,C> + D*u
// B/C come from proj cols [32,48) / [48,64) (uniform across the wave -> bcast)
// ---------------------------------------------------------------------------
__global__ void k_scan(const float* __restrict__ dt,     // (2,8192,512)
                       const float* __restrict__ u,      // (2,8192,512)
                       const float* __restrict__ proj,   // (2,8192,64)
                       const float* __restrict__ A_log,  // (2,512,16)
                       const float* __restrict__ Dp,     // (2,512)
                       float* __restrict__ y) {          // (2,8192,512)
  unsigned i = blockIdx.x * blockDim.x + threadIdx.x;       // 2*8*512 = 8192
  if (i >= (unsigned)(NDIR * BATCH * DI)) return;
  unsigned c   = i & (DI - 1);
  unsigned b   = (i >> 9) & (BATCH - 1);
  unsigned dir = i >> 12;

  const float* dtp = dt   + (size_t)dir * MROWS * DI;
  const float* up  = u    + (size_t)dir * MROWS * DI;
  const float* pp  = proj + (size_t)dir * MROWS * (DTR + 2 * DS);
  float*       yp  = y    + (size_t)dir * MROWS * DI;
  const float* al  = A_log + ((size_t)dir * DI + c) * DS;
  const float  dcoef = Dp[dir * DI + c];

  float A[DS], h[DS];
#pragma unroll
  for (int n = 0; n < DS; ++n) { A[n] = -__expf(al[n]); h[n] = 0.0f; }

  for (int l = 0; l < SEQ; ++l) {
    size_t m   = (size_t)b * SEQ + l;
    size_t rc  = m * DI + c;
    float  dtv = dtp[rc];
    float  uv  = up[rc];
    const float* bv = pp + m * (DTR + 2 * DS) + DTR;        // B: 16 floats
    const float* cv = bv + DS;                              // C: 16 floats
    float  dtu = dtv * uv;
    float  acc = 0.0f;
#pragma unroll
    for (int n = 0; n < DS; ++n) {
      float hn = __expf(dtv * A[n]) * h[n] + dtu * bv[n];
      h[n] = hn;
      acc += hn * cv[n];
    }
    yp[rc] = acc + dcoef * uv;
  }
}

// ---------------------------------------------------------------------------
// Gate: yg = y * silu(z)   (z = xz cols [512,1024)), output f16 for out_proj
// ---------------------------------------------------------------------------
__global__ void k_gate(const float* __restrict__ y, const float* __restrict__ xz,
                       _Float16* __restrict__ yg) {
  unsigned i = blockIdx.x * blockDim.x + threadIdx.x;       // 8192*512
  if (i >= (unsigned)(MROWS * DI)) return;
  unsigned c = i & (DI - 1);
  unsigned m = i >> 9;
  float z = xz[(size_t)m * (2 * DI) + DI + c];
  yg[i] = (_Float16)(y[i] * siluf_(z));
}

// ---------------------------------------------------------------------------
// Combine directions (dir1 un-flipped), scale, write NCHW output.
// ---------------------------------------------------------------------------
__global__ void k_out(const float* __restrict__ od0, const float* __restrict__ od1,
                      const float* __restrict__ scale, float* __restrict__ out) {
  unsigned i = blockIdx.x * blockDim.x + threadIdx.x;       // 8*512*1024
  if (i >= (unsigned)(BATCH * DM * SEQ)) return;
  unsigned l = i & (SEQ - 1);
  unsigned c = (i >> 10) & (DM - 1);
  unsigned b = i >> 19;
  size_t r0 = (size_t)(b * SEQ + l) * DM + c;
  size_t r1 = (size_t)(b * SEQ + (SEQ - 1 - l)) * DM + c;
  out[i] = (od0[r0] + od1[r1]) * scale[0];
}

// ---------------------------------------------------------------------------
extern "C" void kernel_launch(void* const* d_in, const int* in_sizes, int n_in,
                              void* d_out, int out_size, void* d_ws, size_t ws_size,
                              hipStream_t stream) {
  (void)in_sizes; (void)n_in; (void)out_size; (void)ws_size;
  const float* x      = (const float*)d_in[0];   // (8,512,32,32)
  const float* in_w   = (const float*)d_in[1];   // (2,1024,512)
  const float* conv_w = (const float*)d_in[2];   // (2,512,4)
  const float* conv_b = (const float*)d_in[3];   // (2,512)
  const float* xp_w   = (const float*)d_in[4];   // (2,64,512)
  const float* dt_w   = (const float*)d_in[5];   // (2,512,32)
  const float* dt_b   = (const float*)d_in[6];   // (2,512)
  const float* A_log  = (const float*)d_in[7];   // (2,512,16)
  const float* D_p    = (const float*)d_in[8];   // (2,512)
  const float* out_w  = (const float*)d_in[9];   // (2,512,512)
  const float* scale  = (const float*)d_in[10];  // (1,)
  float* out = (float*)d_out;

  // ---- workspace bump allocator (256B aligned) ----
  char* ws = (char*)d_ws;
  size_t off = 0;
  auto bump = [&](size_t bytes) -> char* {
    char* p = ws + off;
    off += (bytes + 255) & ~(size_t)255;
    return p;
  };
  const int NPROJ = DTR + 2 * DS;  // 64
  _Float16* xseq16  = (_Float16*)bump((size_t)NDIR * MROWS * DM * 2);
  _Float16* in_w16  = (_Float16*)bump((size_t)NDIR * 2 * DI * DM * 2);
  _Float16* xp_w16  = (_Float16*)bump((size_t)NDIR * NPROJ * DI * 2);
  _Float16* dt_w16  = (_Float16*)bump((size_t)NDIR * DI * DTR * 2);
  _Float16* out_w16 = (_Float16*)bump((size_t)NDIR * DM * DI * 2);
  float*    xz32    = (float*)bump((size_t)NDIR * MROWS * 2 * DI * 4);
  float*    u32     = (float*)bump((size_t)NDIR * MROWS * DI * 4);
  _Float16* u16     = (_Float16*)bump((size_t)NDIR * MROWS * DI * 2);
  float*    proj32  = (float*)bump((size_t)NDIR * MROWS * NPROJ * 4);
  _Float16* dt16    = (_Float16*)bump((size_t)NDIR * MROWS * DTR * 2);
  float*    dt32    = (float*)bump((size_t)NDIR * MROWS * DI * 4);
  float*    y32     = (float*)bump((size_t)NDIR * MROWS * DI * 4);
  _Float16* yg16    = (_Float16*)bump((size_t)NDIR * MROWS * DI * 2);
  float*    od32    = (float*)bump((size_t)NDIR * MROWS * DM * 4);

  const int T = 256;
  auto blocks = [&](long n) { return (int)((n + T - 1) / T); };
  auto gemm_blocks = [&](int M, int N) { return (M / GBM) * (N / GBN); };

  // weights -> f16
  k_f32_to_f16<<<blocks((long)NDIR * 2 * DI * DM), T, 0, stream>>>(in_w,  in_w16,  NDIR * 2 * DI * DM);
  k_f32_to_f16<<<blocks((long)NDIR * NPROJ * DI), T, 0, stream>>>(xp_w,  xp_w16,  NDIR * NPROJ * DI);
  k_f32_to_f16<<<blocks((long)NDIR * DI * DTR),   T, 0, stream>>>(dt_w,  dt_w16,  NDIR * DI * DTR);
  k_f32_to_f16<<<blocks((long)NDIR * DM * DI),    T, 0, stream>>>(out_w, out_w16, NDIR * DM * DI);

  // both direction sequences (dir1 flipped)
  k_xseq<<<blocks((long)NDIR * MROWS * DM), T, 0, stream>>>(x, xseq16);

  for (int dir = 0; dir < NDIR; ++dir) {
    const size_t dR  = (size_t)dir * MROWS;
    // in_proj: (8192x512) x (1024x512)^T -> (8192x1024)
    k_gemm_wmma<<<gemm_blocks(MROWS, 2 * DI), T, 0, stream>>>(
        xseq16 + dR * DM, in_w16 + (size_t)dir * 2 * DI * DM,
        xz32 + dR * 2 * DI, MROWS, 2 * DI, DM);
    // depthwise conv + silu -> u
    k_conv_silu<<<blocks((long)MROWS * DI), T, 0, stream>>>(
        xz32 + dR * 2 * DI, conv_w + (size_t)dir * DI * DCONV,
        conv_b + (size_t)dir * DI, u32 + dR * DI, u16 + dR * DI);
    // x_proj: (8192x512) x (64x512)^T -> (8192x64)
    k_gemm_wmma<<<gemm_blocks(MROWS, NPROJ), T, 0, stream>>>(
        u16 + dR * DI, xp_w16 + (size_t)dir * NPROJ * DI,
        proj32 + dR * NPROJ, MROWS, NPROJ, DI);
    // dt path: split, project, softplus
    k_dtsplit<<<blocks((long)MROWS * DTR), T, 0, stream>>>(
        proj32 + dR * NPROJ, dt16 + dR * DTR);
    k_gemm_wmma<<<gemm_blocks(MROWS, DI), T, 0, stream>>>(
        dt16 + dR * DTR, dt_w16 + (size_t)dir * DI * DTR,
        dt32 + dR * DI, MROWS, DI, DTR);
    k_softplus<<<blocks((long)MROWS * DI), T, 0, stream>>>(
        dt32 + dR * DI, dt_b + (size_t)dir * DI);
  }

  // selective scan (both directions)
  k_scan<<<blocks((long)NDIR * BATCH * DI), T, 0, stream>>>(
      dt32, u32, proj32, A_log, D_p, y32);

  for (int dir = 0; dir < NDIR; ++dir) {
    const size_t dR = (size_t)dir * MROWS;
    k_gate<<<blocks((long)MROWS * DI), T, 0, stream>>>(
        y32 + dR * DI, xz32 + dR * 2 * DI, yg16 + dR * DI);
    // out_proj: (8192x512) x (512x512)^T -> (8192x512)
    k_gemm_wmma<<<gemm_blocks(MROWS, DM), T, 0, stream>>>(
        yg16 + dR * DI, out_w16 + (size_t)dir * DM * DI,
        od32 + dR * DM, MROWS, DM, DI);
  }

  // sum directions (dir1 re-flipped), scale, NCHW
  k_out<<<blocks((long)BATCH * DM * SEQ), T, 0, stream>>>(
      od32, od32 + (size_t)MROWS * DM, scale, out);
}
